// DeformableTransformer_84095459656058
// MI455X (gfx1250) — compile-verified, matrix operand order checked by reading, not run on hardware
//
#include <hip/hip_runtime.h>

// ---------------------------------------------------------------------------
// Deformable transformer decoder (6 layers) for MI455X / gfx1250.
// GEMMs: v_wmma_f32_16x16x32_f16, fp16-resident operands (weights/src
// converted once; every producer writes an f16 shadow for GEMM consumers),
// double-buffered LDS staging via global_load_async_to_lds_b128 (ASYNCcnt)
// when the toolchain exposes the builtin, b128 copies otherwise.
// ---------------------------------------------------------------------------

typedef _Float16 v4h  __attribute__((ext_vector_type(4)));
typedef _Float16 v8h  __attribute__((ext_vector_type(8)));
typedef _Float16 v16h __attribute__((ext_vector_type(16)));
typedef float    v8f  __attribute__((ext_vector_type(8)));
typedef int      v4i  __attribute__((vector_size(16)));

#define DT_B       16
#define DT_LQ      300
#define DT_D       512
#define DT_H       8
#define DT_DH      64
#define DT_LEN_IN  3840
#define DT_FFN     2048
#define DT_HLP     128
#define DT_LAYERS  6
#define SC_LD16    304            // padded f16 pitch for scores (16B-aligned rows)

#if defined(__has_builtin)
#if __has_builtin(__builtin_amdgcn_global_load_async_to_lds_b128) && \
    __has_builtin(__builtin_amdgcn_s_wait_asynccnt)
#define USE_ASYNC_LDS 1
#endif
#endif
#ifndef USE_ASYNC_LDS
#define USE_ASYNC_LDS 0
#endif

__device__ __forceinline__ void lds_copy16(const _Float16* g, _Float16* l)
{
#if USE_ASYNC_LDS
    __builtin_amdgcn_global_load_async_to_lds_b128(
        (__attribute__((address_space(1))) v4i*)g,
        (__attribute__((address_space(3))) v4i*)l, 0, 0);
#else
    *(v8h*)l = *(const v8h*)g;
#endif
}

__device__ __forceinline__ void wait_async_lds()
{
#if USE_ASYNC_LDS
    __builtin_amdgcn_s_wait_asynccnt(0);
#endif
}

// ---------------------------------------------------------------------------
// WMMA GEMM with fp16 operands:
//   C[m,n] = sum_k A[m,k]*B[k,n] (+bias[n]) (+relu)
//   BT=true : B[k,n] = B16[n*ldb + k]   (weight layout (N,K))
//   BT=false: B[k,n] = B16[k*ldb + n]
// Optional f32 output C and/or f16 shadow C16 (same ldc / z strides).
// Tile 64x64, BK=32, 128 threads (4 waves), double-buffered LDS.
// ---------------------------------------------------------------------------
template <bool BT, bool RELU>
__global__ void __launch_bounds__(128)
gemm_wmma_kernel(const _Float16* __restrict__ A16, int lda, long sA0, long sA1,
                 const _Float16* __restrict__ B16, int ldb, long sB0, long sB1,
                 float* __restrict__ C, _Float16* __restrict__ C16,
                 int ldc, long sC0, long sC1,
                 const float* __restrict__ bias,
                 int M, int N, int K, int zdiv)
{
    __shared__ _Float16 Ah[2][64][40];   // row stride 40 halves => 16B-aligned rows
    __shared__ _Float16 Bh[2][64][40];   // n-major over k

    const int z  = blockIdx.z;
    const int z0 = z / zdiv, z1 = z - z0 * zdiv;
    A16 += (long)z0 * sA0 + (long)z1 * sA1;
    B16 += (long)z0 * sB0 + (long)z1 * sB1;
    const long coff = (long)z0 * sC0 + (long)z1 * sC1;

    const int n0   = blockIdx.x * 64;
    const int m0   = blockIdx.y * 64;
    const int tid  = threadIdx.x;
    const int wave = tid >> 5;
    const int lane = tid & 31;
    const int fr   = lane & 15;
    const int kb   = (lane >> 4) * 8;

    const bool mfull = (m0 + 64 <= M);
    const bool nfull = (n0 + 64 <= N);

    // --- stage A tile 64x32 halves ---
    auto stageA = [&](int buf, int kk) {
        if (mfull && (kk + 32 <= K)) {
            const int r0 = tid >> 2;           // 0..31
            const int c8 = (tid & 3) * 8;      // 0,8,16,24
#pragma unroll
            for (int p = 0; p < 2; ++p) {
                const int r = r0 + p * 32;
                lds_copy16(&A16[(long)(m0 + r) * lda + kk + c8], &Ah[buf][r][c8]);
            }
        } else {
            for (int e = tid; e < 64 * 32; e += 128) {
                const int r = e >> 5, c = e & 31;
                const int gm = m0 + r, gk = kk + c;
                const int cm = gm < M ? gm : M - 1;
                const int ck = gk < K ? gk : K - 1;
                const _Float16 v = A16[(long)cm * lda + ck];      // always in-bounds
                Ah[buf][r][c] = (gm < M && gk < K) ? v : (_Float16)0.0f;
            }
        }
    };

    // --- stage B tile (Bh[n_local][k_local]) ---
    auto stageB = [&](int buf, int kk) {
        if (BT) {
            if (nfull && (kk + 32 <= K)) {
                const int r0 = tid >> 2;
                const int c8 = (tid & 3) * 8;
#pragma unroll
                for (int p = 0; p < 2; ++p) {
                    const int r = r0 + p * 32;
                    lds_copy16(&B16[(long)(n0 + r) * ldb + kk + c8], &Bh[buf][r][c8]);
                }
            } else {
                for (int e = tid; e < 64 * 32; e += 128) {
                    const int r = e >> 5, c = e & 31;
                    const int gn = n0 + r, gk = kk + c;
                    const int cn = gn < N ? gn : N - 1;
                    const int ck = gk < K ? gk : K - 1;
                    const _Float16 v = B16[(long)cn * ldb + ck];
                    Bh[buf][r][c] = (gn < N && gk < K) ? v : (_Float16)0.0f;
                }
            }
        } else {
            if (nfull && (kk + 32 <= K)) {
                const int ng = (tid & 7) * 8;      // n offset 0..56
                const int k0 = tid >> 3;           // 0..15
#pragma unroll
                for (int p = 0; p < 2; ++p) {
                    const int k = k0 + p * 16;
                    const v8h v = *(const v8h*)&B16[(long)(kk + k) * ldb + n0 + ng];
#pragma unroll
                    for (int j = 0; j < 8; ++j) Bh[buf][ng + j][k] = v[j];
                }
            } else {
                for (int e = tid; e < 64 * 32; e += 128) {
                    const int n = e & 63, k = e >> 6;   // n fastest -> coalesced
                    const int gn = n0 + n, gk = kk + k;
                    const int cn = gn < N ? gn : N - 1;
                    const int ck = gk < K ? gk : K - 1;
                    const _Float16 v = B16[(long)ck * ldb + cn];
                    Bh[buf][n][k] = (gn < N && gk < K) ? v : (_Float16)0.0f;
                }
            }
        }
    };

    v8f acc[4] = {v8f{}, v8f{}, v8f{}, v8f{}};

    const int ksteps = (K + 31) >> 5;

    stageA(0, 0);
    stageB(0, 0);
    wait_async_lds();
    __syncthreads();

    int buf = 0;
    for (int ks = 0; ks < ksteps; ++ks) {
        if (ks + 1 < ksteps) {                   // prefetch next tile
            stageA(buf ^ 1, (ks + 1) << 5);
            stageB(buf ^ 1, (ks + 1) << 5);
        }

        const int mrow = wave * 16;
        v8h alo = *(const v8h*)&Ah[buf][mrow + fr][kb];
        v8h ahi = *(const v8h*)&Ah[buf][mrow + fr][kb + 16];
        v16h af;
#pragma unroll
        for (int i = 0; i < 8; ++i) { af[i] = alo[i]; af[8 + i] = ahi[i]; }

#pragma unroll
        for (int nt = 0; nt < 4; ++nt) {
            v8h blo = *(const v8h*)&Bh[buf][nt * 16 + fr][kb];
            v8h bhi = *(const v8h*)&Bh[buf][nt * 16 + fr][kb + 16];
            v16h bf;
#pragma unroll
            for (int i = 0; i < 8; ++i) { bf[i] = blo[i]; bf[8 + i] = bhi[i]; }
            acc[nt] = __builtin_amdgcn_wmma_f32_16x16x32_f16(
                false, af, false, bf, (short)0, acc[nt], false, false);
        }

        wait_async_lds();     // loads issued this iter done (overlapped with wmma)
        __syncthreads();
        buf ^= 1;
    }

    // ---- epilogue ----
#pragma unroll
    for (int nt = 0; nt < 4; ++nt) {
        const int gn = n0 + nt * 16 + fr;
        const bool nok = (gn < N);
        const float bv = (bias && nok) ? bias[gn] : 0.0f;
#pragma unroll
        for (int v = 0; v < 8; ++v) {
            const int gm = m0 + wave * 16 + v + ((lane >> 4) * 8);
            if (gm < M && nok) {
                float r = acc[nt][v] + bv;
                if (RELU) r = fmaxf(r, 0.0f);
                const long o = coff + (long)gm * ldc + gn;
                if (C)   C[o]   = r;
                if (C16) C16[o] = (_Float16)r;
            }
        }
    }
}

// ---------------------------------------------------------------------------
// f32 -> f16 conversion (n multiple of 4)
// ---------------------------------------------------------------------------
__global__ void __launch_bounds__(256)
cvt_f16_kernel(_Float16* __restrict__ dst, const float* __restrict__ src, long n)
{
    long i = ((long)blockIdx.x * 256 + threadIdx.x) * 4;
    if (i < n) {
        const float4 v = *(const float4*)&src[i];
        v4h h;
        h[0] = (_Float16)v.x; h[1] = (_Float16)v.y;
        h[2] = (_Float16)v.z; h[3] = (_Float16)v.w;
        *(v4h*)&dst[i] = h;
    }
}

// ---------------------------------------------------------------------------
// o16 = (f16)(a + b)
// ---------------------------------------------------------------------------
__global__ void __launch_bounds__(256)
add2_f16_kernel(_Float16* __restrict__ o16, const float* __restrict__ a,
                const float* __restrict__ b, long n)
{
    long i = (long)blockIdx.x * 256 + threadIdx.x;
    if (i < n) o16[i] = (_Float16)(a[i] + b[i]);
}

// ---------------------------------------------------------------------------
// Row softmax of scale*x; writes f16 shadow with pitch ld16.
// ---------------------------------------------------------------------------
__global__ void __launch_bounds__(256)
softmax_rows_kernel(float* __restrict__ x, _Float16* __restrict__ x16,
                    int cols, int ld16, float scale)
{
    __shared__ float sh[256];
    const long row = blockIdx.x;
    float* xr = x + row * (long)cols;
    _Float16* xo = x16 + row * (long)ld16;
    const int t = threadIdx.x;

    float mx = -1e30f;
    for (int c = t; c < cols; c += 256) mx = fmaxf(mx, scale * xr[c]);
    sh[t] = mx; __syncthreads();
    for (int s = 128; s > 0; s >>= 1) {
        if (t < s) sh[t] = fmaxf(sh[t], sh[t + s]);
        __syncthreads();
    }
    mx = sh[0]; __syncthreads();

    float sum = 0.0f;
    for (int c = t; c < cols; c += 256) {
        float e = __expf(scale * xr[c] - mx);
        xr[c] = e; sum += e;
    }
    sh[t] = sum; __syncthreads();
    for (int s = 128; s > 0; s >>= 1) {
        if (t < s) sh[t] += sh[t + s];
        __syncthreads();
    }
    const float inv = 1.0f / sh[0];
    for (int c = t; c < cols; c += 256) xo[c] = (_Float16)(xr[c] * inv);
}

// ---------------------------------------------------------------------------
// out[row] = LN(out[row] + t2[row]); also writes f16 shadow o16.
// ---------------------------------------------------------------------------
__global__ void __launch_bounds__(256)
add_ln_kernel(float* __restrict__ out, _Float16* __restrict__ out16,
              const float* __restrict__ t2,
              const float* __restrict__ g, const float* __restrict__ bb)
{
    __shared__ float sh[256];
    const long row = blockIdx.x;
    float* o = out + row * (long)DT_D;
    _Float16* o16 = out16 + row * (long)DT_D;
    const float* a = t2 + row * (long)DT_D;
    const int t = threadIdx.x;

    float x0 = o[t] + a[t];
    float x1 = o[t + 256] + a[t + 256];

    sh[t] = x0 + x1; __syncthreads();
    for (int s = 128; s > 0; s >>= 1) {
        if (t < s) sh[t] += sh[t + s];
        __syncthreads();
    }
    const float mean = sh[0] * (1.0f / (float)DT_D);
    __syncthreads();

    const float d0 = x0 - mean, d1 = x1 - mean;
    sh[t] = d0 * d0 + d1 * d1; __syncthreads();
    for (int s = 128; s > 0; s >>= 1) {
        if (t < s) sh[t] += sh[t + s];
        __syncthreads();
    }
    const float inv = rsqrtf(sh[0] * (1.0f / (float)DT_D) + 1e-5f);
    const float r0 = d0 * inv * g[t]       + bb[t];
    const float r1 = d1 * inv * g[t + 256] + bb[t + 256];
    o[t] = r0;         o16[t] = (_Float16)r0;
    o[t + 256] = r1;   o16[t + 256] = (_Float16)r1;
}

// ---------------------------------------------------------------------------
// Attention-weight softmax over last 16 per (b,q,h), in place (f32).
// ---------------------------------------------------------------------------
__global__ void __launch_bounds__(256)
aw_softmax_kernel(float* __restrict__ aw, int nrows)
{
    int i = blockIdx.x * 256 + threadIdx.x;
    if (i >= nrows) return;
    float* r = aw + (long)i * 16;
    float mx = -1e30f;
#pragma unroll
    for (int j = 0; j < 16; ++j) mx = fmaxf(mx, r[j]);
    float s = 0.0f;
#pragma unroll
    for (int j = 0; j < 16; ++j) { float e = __expf(r[j] - mx); r[j] = e; s += e; }
    const float inv = 1.0f / s;
#pragma unroll
    for (int j = 0; j < 16; ++j) r[j] *= inv;
}

// ---------------------------------------------------------------------------
// Deformable sampling -> f16 shadow (consumed only by the output-proj GEMM).
// value stays f32 and is L2-resident (126 MB < 192 MB L2).
// ---------------------------------------------------------------------------
__global__ void __launch_bounds__(64)
deform_sample_kernel(const float* __restrict__ value,
                     const float* __restrict__ offb,
                     const float* __restrict__ awb,
                     const float* __restrict__ refp,
                     const float* __restrict__ svr,
                     _Float16* __restrict__ sampled16)
{
    const int idx = blockIdx.x;
    const int h = idx & 7;
    const int q = (idx >> 3) % DT_LQ;
    const int b = idx / (DT_H * DT_LQ);
    const int d = threadIdx.x;

    const int Ts[4] = {2048, 1024, 512, 256};
    const int Ss[4] = {0, 2048, 3072, 3584};

    const float rp = refp[b * DT_LQ + q];
    const long qbase = ((long)(b * DT_LQ + q) * DT_H + h) * 16;
    const float* vb = value + ((long)b * DT_LEN_IN) * DT_D + h * DT_DH + d;

    float acc = 0.0f;
#pragma unroll
    for (int l = 0; l < 4; ++l) {
        const int T = Ts[l], st = Ss[l];
        const float ref = rp * svr[b * 4 + l];
#pragma unroll
        for (int p = 0; p < 4; ++p) {
            const float off = offb[qbase + l * 4 + p];
            const float a   = awb[qbase + l * 4 + p];
            const float x   = ref * (float)T + off - 0.5f;
            const float x0  = floorf(x);
            const float fr  = x - x0;
            const int i0 = (int)fminf(fmaxf(x0, 0.0f), (float)(T - 1));
            const int i1 = (int)fminf(fmaxf(x0 + 1.0f, 0.0f), (float)(T - 1));
            const float m0 = (x0 >= 0.0f && x0 <= (float)(T - 1)) ? 1.0f : 0.0f;
            const float m1 = (x0 + 1.0f >= 0.0f && x0 + 1.0f <= (float)(T - 1)) ? 1.0f : 0.0f;
            const float g0 = vb[(long)(st + i0) * DT_D];
            const float g1 = vb[(long)(st + i1) * DT_D];
            acc += a * (g0 * m0 * (1.0f - fr) + g1 * m1 * fr);
        }
    }
    sampled16[(long)(b * DT_LQ + q) * DT_D + h * DT_DH + d] = (_Float16)acc;
}

// ---------------------------------------------------------------------------
// Host orchestration
// ---------------------------------------------------------------------------
static inline void launch_gemm(const _Float16* A, int lda, long sA0, long sA1,
                               const _Float16* Bp, int ldb, long sB0, long sB1,
                               float* C, _Float16* C16, int ldc, long sC0, long sC1,
                               const float* bias, int M, int N, int K,
                               int Z, int zdiv, bool bt, bool relu,
                               hipStream_t stream)
{
    dim3 grid((N + 63) / 64, (M + 63) / 64, Z);
    dim3 blk(128);
    if (bt) {
        if (relu)
            gemm_wmma_kernel<true, true><<<grid, blk, 0, stream>>>(
                A, lda, sA0, sA1, Bp, ldb, sB0, sB1, C, C16, ldc, sC0, sC1, bias, M, N, K, zdiv);
        else
            gemm_wmma_kernel<true, false><<<grid, blk, 0, stream>>>(
                A, lda, sA0, sA1, Bp, ldb, sB0, sB1, C, C16, ldc, sC0, sC1, bias, M, N, K, zdiv);
    } else {
        if (relu)
            gemm_wmma_kernel<false, true><<<grid, blk, 0, stream>>>(
                A, lda, sA0, sA1, Bp, ldb, sB0, sB1, C, C16, ldc, sC0, sC1, bias, M, N, K, zdiv);
        else
            gemm_wmma_kernel<false, false><<<grid, blk, 0, stream>>>(
                A, lda, sA0, sA1, Bp, ldb, sB0, sB1, C, C16, ldc, sC0, sC1, bias, M, N, K, zdiv);
    }
}

static inline void cvt16(_Float16* dst, const float* src, long n, hipStream_t s)
{
    cvt_f16_kernel<<<dim3((unsigned)((n / 4 + 255) / 256)), dim3(256), 0, s>>>(dst, src, n);
}

extern "C" void kernel_launch(void* const* d_in, const int* in_sizes, int n_in,
                              void* d_out, int out_size, void* d_ws, size_t ws_size,
                              hipStream_t stream)
{
    (void)in_sizes; (void)n_in; (void)out_size; (void)ws_size;

    const float* tgt   = (const float*)d_in[0];
    const float* refp  = (const float*)d_in[1];
    const float* src   = (const float*)d_in[2];
    const float* svr   = (const float*)d_in[3];
    const float* qpos  = (const float*)d_in[4];
    const float* sa_in_w  = (const float*)d_in[5];
    const float* sa_in_b  = (const float*)d_in[6];
    const float* sa_out_w = (const float*)d_in[7];
    const float* sa_out_b = (const float*)d_in[8];
    const float* off_w = (const float*)d_in[9];
    const float* off_b = (const float*)d_in[10];
    const float* aw_w  = (const float*)d_in[11];
    const float* aw_b  = (const float*)d_in[12];
    const float* val_w = (const float*)d_in[13];
    const float* val_b = (const float*)d_in[14];
    const float* outp_w = (const float*)d_in[15];
    const float* outp_b = (const float*)d_in[16];
    const float* ffn_w1 = (const float*)d_in[17];
    const float* ffn_b1 = (const float*)d_in[18];
    const float* ffn_w2 = (const float*)d_in[19];
    const float* ffn_b2 = (const float*)d_in[20];
    const float* ln1_g = (const float*)d_in[21];
    const float* ln1_b = (const float*)d_in[22];
    const float* ln2_g = (const float*)d_in[23];
    const float* ln2_b = (const float*)d_in[24];
    const float* ln3_g = (const float*)d_in[25];
    const float* ln3_b = (const float*)d_in[26];

    const long NQ   = (long)DT_B * DT_LQ;            // 4800
    const long NQD  = NQ * DT_D;                     // 2,457,600
    const long NV   = (long)DT_B * DT_LEN_IN;        // 61,440
    const long NSC  = (long)DT_B * DT_H * DT_LQ;     // 38,400 score rows

    // ---- f32 workspace ----
    float* w = (float*)d_ws;
    float* out    = w;  w += NQD;
    float* scores = w;  w += NSC * DT_LQ;
    float* t2     = w;  w += NQD;
    float* value  = w;  w += NV * DT_D;
    float* offbuf = w;  w += NQ * DT_HLP;
    float* awbuf  = w;  w += NQ * DT_HLP;

    // ---- f16 workspace ----
    _Float16* h = (_Float16*)w;
    _Float16* out16     = h;  h += NQD;
    _Float16* qbuf16    = h;  h += NQD;
    _Float16* qkv16     = h;  h += NQ * 3 * DT_D;
    _Float16* scores16  = h;  h += NSC * SC_LD16;
    _Float16* attnout16 = h;  h += NQD;
    _Float16* sampled16 = h;  h += NQD;
    _Float16* ffnh16    = h;  h += NQ * DT_FFN;
    _Float16* src16     = h;  h += NV * DT_D;
    _Float16* w16_in    = h;  h += (long)DT_LAYERS * 3 * DT_D * DT_D;
    _Float16* w16_sao   = h;  h += (long)DT_LAYERS * DT_D * DT_D;
    _Float16* w16_off   = h;  h += (long)DT_LAYERS * DT_HLP * DT_D;
    _Float16* w16_aw    = h;  h += (long)DT_LAYERS * DT_HLP * DT_D;
    _Float16* w16_val   = h;  h += (long)DT_LAYERS * DT_D * DT_D;
    _Float16* w16_outp  = h;  h += (long)DT_LAYERS * DT_D * DT_D;
    _Float16* w16_f1    = h;  h += (long)DT_LAYERS * DT_FFN * DT_D;
    _Float16* w16_f2    = h;  h += (long)DT_LAYERS * DT_D * DT_FFN;

    // ---- one-time conversions ----
    cvt16(src16,    src,      NV * DT_D, stream);
    cvt16(w16_in,   sa_in_w,  (long)DT_LAYERS * 3 * DT_D * DT_D, stream);
    cvt16(w16_sao,  sa_out_w, (long)DT_LAYERS * DT_D * DT_D, stream);
    cvt16(w16_off,  off_w,    (long)DT_LAYERS * DT_HLP * DT_D, stream);
    cvt16(w16_aw,   aw_w,     (long)DT_LAYERS * DT_HLP * DT_D, stream);
    cvt16(w16_val,  val_w,    (long)DT_LAYERS * DT_D * DT_D, stream);
    cvt16(w16_outp, outp_w,   (long)DT_LAYERS * DT_D * DT_D, stream);
    cvt16(w16_f1,   ffn_w1,   (long)DT_LAYERS * DT_FFN * DT_D, stream);
    cvt16(w16_f2,   ffn_w2,   (long)DT_LAYERS * DT_D * DT_FFN, stream);

    (void)hipMemcpyAsync(out, tgt, NQD * sizeof(float), hipMemcpyDeviceToDevice, stream);
    cvt16(out16, tgt, NQD, stream);

    const long sBH_A = (long)DT_LQ * 1536;            // per-batch stride in qkv16
    const long sBH_S16 = (long)DT_LQ * SC_LD16;       // per-head stride in scores16
    const long sBH_S = (long)DT_H * DT_LQ * DT_LQ;    // per-batch stride in scores f32

    for (int lid = 0; lid < DT_LAYERS; ++lid) {
        const _Float16* hw_in  = w16_in  + (long)lid * 3 * DT_D * DT_D;
        const _Float16* hw_sao = w16_sao + (long)lid * DT_D * DT_D;
        const _Float16* hw_off = w16_off + (long)lid * DT_HLP * DT_D;
        const _Float16* hw_aw  = w16_aw  + (long)lid * DT_HLP * DT_D;
        const _Float16* hw_val = w16_val + (long)lid * DT_D * DT_D;
        const _Float16* hw_out = w16_outp + (long)lid * DT_D * DT_D;
        const _Float16* hw_f1  = w16_f1  + (long)lid * DT_FFN * DT_D;
        const _Float16* hw_f2  = w16_f2  + (long)lid * DT_D * DT_FFN;
        const float* b_in  = sa_in_b  + (long)lid * 3 * DT_D;
        const float* b_sao = sa_out_b + (long)lid * DT_D;
        const float* b_off = off_b + (long)lid * DT_HLP;
        const float* b_aw  = aw_b  + (long)lid * DT_HLP;
        const float* b_val = val_b + (long)lid * DT_D;
        const float* b_out = outp_b + (long)lid * DT_D;
        const float* b_f1  = ffn_b1 + (long)lid * DT_FFN;
        const float* b_f2  = ffn_b2 + (long)lid * DT_D;

        // ---- self-attention ----
        add2_f16_kernel<<<dim3((NQD + 255) / 256), dim3(256), 0, stream>>>(qbuf16, out, qpos, NQD);
        // q,k projections (f16 shadow only)
        launch_gemm(qbuf16, DT_D, 0, 0, hw_in, DT_D, 0, 0,
                    nullptr, qkv16, 1536, 0, 0, b_in, (int)NQ, 1024, DT_D, 1, 1, true, false, stream);
        // v projection (f16 shadow only)
        launch_gemm(out16, DT_D, 0, 0, hw_in + (long)1024 * DT_D, DT_D, 0, 0,
                    nullptr, qkv16 + 1024, 1536, 0, 0, b_in + 1024, (int)NQ, DT_D, DT_D, 1, 1, true, false, stream);
        // scores[b,h] = Q K^T (f32)
        launch_gemm(qkv16, 1536, sBH_A, 64,
                    qkv16 + 512, 1536, sBH_A, 64,
                    scores, nullptr, DT_LQ, sBH_S, (long)DT_LQ * DT_LQ,
                    nullptr, DT_LQ, DT_LQ, DT_DH,
                    DT_B * DT_H, DT_H, true, false, stream);
        softmax_rows_kernel<<<dim3((unsigned)NSC), dim3(256), 0, stream>>>(
            scores, scores16, DT_LQ, SC_LD16, 0.125f);
        // attnout16[b,:,h,:] = attn @ V
        launch_gemm(scores16, SC_LD16, (long)DT_H * sBH_S16, sBH_S16,
                    qkv16 + 1024, 1536, sBH_A, 64,
                    nullptr, attnout16, DT_D, (long)DT_LQ * DT_D, 64,
                    nullptr, DT_LQ, DT_DH, DT_LQ,
                    DT_B * DT_H, DT_H, false, false, stream);
        launch_gemm(attnout16, DT_D, 0, 0, hw_sao, DT_D, 0, 0,
                    t2, nullptr, DT_D, 0, 0, b_sao, (int)NQ, DT_D, DT_D, 1, 1, true, false, stream);
        add_ln_kernel<<<dim3((int)NQ), dim3(256), 0, stream>>>(
            out, out16, t2, ln2_g + lid * DT_D, ln2_b + lid * DT_D);

        // ---- deformable cross-attention ----
        launch_gemm(src16, DT_D, 0, 0, hw_val, DT_D, 0, 0,
                    value, nullptr, DT_D, 0, 0, b_val, (int)NV, DT_D, DT_D, 1, 1, true, false, stream);
        add2_f16_kernel<<<dim3((NQD + 255) / 256), dim3(256), 0, stream>>>(qbuf16, out, qpos, NQD);
        launch_gemm(qbuf16, DT_D, 0, 0, hw_off, DT_D, 0, 0,
                    offbuf, nullptr, DT_HLP, 0, 0, b_off, (int)NQ, DT_HLP, DT_D, 1, 1, true, false, stream);
        launch_gemm(qbuf16, DT_D, 0, 0, hw_aw, DT_D, 0, 0,
                    awbuf, nullptr, DT_HLP, 0, 0, b_aw, (int)NQ, DT_HLP, DT_D, 1, 1, true, false, stream);
        aw_softmax_kernel<<<dim3((DT_B * DT_LQ * DT_H + 255) / 256), dim3(256), 0, stream>>>(
            awbuf, DT_B * DT_LQ * DT_H);
        deform_sample_kernel<<<dim3(DT_B * DT_LQ * DT_H), dim3(64), 0, stream>>>(
            value, offbuf, awbuf, refp, svr, sampled16);
        launch_gemm(sampled16, DT_D, 0, 0, hw_out, DT_D, 0, 0,
                    t2, nullptr, DT_D, 0, 0, b_out, (int)NQ, DT_D, DT_D, 1, 1, true, false, stream);
        add_ln_kernel<<<dim3((int)NQ), dim3(256), 0, stream>>>(
            out, out16, t2, ln1_g + lid * DT_D, ln1_b + lid * DT_D);

        // ---- FFN ----
        launch_gemm(out16, DT_D, 0, 0, hw_f1, DT_D, 0, 0,
                    nullptr, ffnh16, DT_FFN, 0, 0, b_f1, (int)NQ, DT_FFN, DT_D, 1, 1, true, true, stream);
        launch_gemm(ffnh16, DT_FFN, 0, 0, hw_f2, DT_FFN, 0, 0,
                    t2, nullptr, DT_D, 0, 0, b_f2, (int)NQ, DT_D, DT_FFN, 1, 1, true, false, stream);
        add_ln_kernel<<<dim3((int)NQ), dim3(256), 0, stream>>>(
            out, out16, t2, ln3_g + lid * DT_D, ln3_b + lid * DT_D);
    }

    (void)hipMemcpyAsync(d_out, out, NQD * sizeof(float), hipMemcpyDeviceToDevice, stream);
}
